// SimpleGlossFreeModel_42150809043141
// MI455X (gfx1250) — matrix operand order repaired and checked by reading, not compiled, not run on hardware
//
#include <hip/hip_runtime.h>
#include <hip/hip_bf16.h>

typedef __attribute__((ext_vector_type(16))) __bf16 v16bf;
typedef __attribute__((ext_vector_type(8)))  float  v8f;
typedef __attribute__((ext_vector_type(4)))  int    v4i;

// ---------- helpers ----------
__device__ __forceinline__ unsigned short f2bf(float f) {
    union { float f; unsigned u; } c; c.f = f;
    unsigned u = c.u;
    unsigned r = (u + 0x7FFFu + ((u >> 16) & 1u)) >> 16; // RNE
    return (unsigned short)r;
}

// ---------- CDNA5 async global->LDS staging (guarded) ----------
#if defined(__has_builtin)
#  if __has_builtin(__builtin_amdgcn_global_load_async_to_lds_b128)
#    define HAVE_ASYNC_LDS 1
#  endif
#endif

#ifdef HAVE_ASYNC_LDS
typedef __attribute__((address_space(1))) v4i* as1_v4i;
typedef __attribute__((address_space(3))) v4i* as3_v4i;
#endif

__device__ __forceinline__ void cp16_g2s(const unsigned short* g, unsigned short* s) {
#ifdef HAVE_ASYNC_LDS
    __builtin_amdgcn_global_load_async_to_lds_b128(
        (as1_v4i)(v4i*)(unsigned short*)g,
        (as3_v4i)(v4i*)s, 0, 0);
#else
    *(uint4*)s = *(const uint4*)g;
#endif
}

__device__ __forceinline__ void wait_stage() {
#ifdef HAVE_ASYNC_LDS
#  if __has_builtin(__builtin_amdgcn_s_wait_asynccnt)
    __builtin_amdgcn_s_wait_asynccnt(0);
#  else
    asm volatile("s_wait_asynccnt 0" ::: "memory");
#  endif
#endif
}

// ---------- problem constants ----------
#define MB   64            // batch
#define TT   512           // time
#define MM   (MB*TT)       // 32768 rows
#define DD   1024
#define H1   768
#define H2   512
#define KC   512           // codebook entries
#define P1   256
#define P2   128

// ---------- WMMA GEMM: C[MxN] f32 = A[MxK] bf16 * BT[NxK] bf16 ----------
// B is supplied TRANSPOSED (N-major), so both staging streams are contiguous.
#define BM 128
#define BN 64
#define BK 32
#define BKP 40   // padded LDS K-stride; *2 bytes = 80 (multiple of 16)

__global__ __launch_bounds__(256) void gemm_bf16_wmma(
    const unsigned short* __restrict__ A, const unsigned short* __restrict__ BT,
    float* __restrict__ C, int M, int N, int K)
{
    __shared__ unsigned short lsA[2][BM * BKP];
    __shared__ unsigned short lsB[2][BN * BKP];   // [n][k]

    const int tid   = threadIdx.x;
    const int lane  = tid & 31;
    const int wave  = tid >> 5;
    const int waveM = wave & 3;     // 4 waves along M, 32 rows each
    const int waveN = wave >> 2;    // 2 waves along N, 32 cols each
    const size_t blockRow = (size_t)blockIdx.y * BM;
    const size_t blockCol = (size_t)blockIdx.x * BN;

    // staging: 16B chunks. A has 512 chunks (2/thread), B has 256 (1/thread).
    const int aRow0 = tid >> 2;               // 0..63
    const int aRow1 = aRow0 + 64;             // 64..127
    const int aCol  = (tid & 3) * 8;          // 0,8,16,24
    const int bRow  = tid >> 2;               // n-index 0..63
    const int bCol  = (tid & 3) * 8;

    const unsigned short* gA0 = A  + (blockRow + aRow0) * K + aCol;
    const unsigned short* gA1 = A  + (blockRow + aRow1) * K + aCol;
    const unsigned short* gB  = BT + (blockCol + bRow)  * K + bCol;

    unsigned short* sA0 = &lsA[0][aRow0 * BKP + aCol];
    unsigned short* sA1 = &lsA[0][aRow1 * BKP + aCol];
    unsigned short* sB  = &lsB[0][bRow  * BKP + bCol];
    const int strideA = BM * BKP;   // elements between buffers
    const int strideB = BN * BKP;

    // fragment base pointers (both LDS buffers)
    const int r     = lane & 15;
    const int half  = lane >> 4;
    const int koff  = half * 8;    // A: lanes 0-15 K0..7/16..23, 16-31 K8..15/24..31
    const int khalf = half * 16;   // unused; B is read with koff, same k order as A
    const int rowBase = waveM * 32;
    const int colBase = waveN * 32;
    const unsigned short* fA0[2]; const unsigned short* fA1[2];
    const unsigned short* fB0[2]; const unsigned short* fB1[2];
    #pragma unroll
    for (int b = 0; b < 2; ++b) {
        fA0[b] = &lsA[b][(rowBase + r)      * BKP];
        fA1[b] = &lsA[b][(rowBase + 16 + r) * BKP];
        fB0[b] = &lsB[b][(colBase + r)      * BKP + koff];
        fB1[b] = &lsB[b][(colBase + 16 + r) * BKP + koff];
    }

    v8f acc00 = {}, acc01 = {}, acc10 = {}, acc11 = {};
    const int KT = K / BK;

    // prologue stage into buffer 0
    cp16_g2s(gA0, sA0);
    cp16_g2s(gA1, sA1);
    cp16_g2s(gB,  sB);
    gA0 += BK; gA1 += BK; gB += BK;
    wait_stage();
    __syncthreads();

    for (int kt = 0; kt < KT; ++kt) {
        const int cur = kt & 1;
        if (kt + 1 < KT) {
            const int nb = (cur ^ 1);
            cp16_g2s(gA0, sA0 + nb * strideA);
            cp16_g2s(gA1, sA1 + nb * strideA);
            cp16_g2s(gB,  sB  + nb * strideB);
            gA0 += BK; gA1 += BK; gB += BK;
        }

        union { uint4 q[2]; v16bf v; } a0, a1, b0, b1;
        a0.q[0] = *(const uint4*)(fA0[cur] + koff);
        a0.q[1] = *(const uint4*)(fA0[cur] + 16 + koff);
        a1.q[0] = *(const uint4*)(fA1[cur] + koff);
        a1.q[1] = *(const uint4*)(fA1[cur] + 16 + koff);
        b0.q[0] = *(const uint4*)(fB0[cur]);
        b0.q[1] = *(const uint4*)(fB0[cur] + 16);
        b1.q[0] = *(const uint4*)(fB1[cur]);
        b1.q[1] = *(const uint4*)(fB1[cur] + 16);

        acc00 = __builtin_amdgcn_wmma_f32_16x16x32_bf16(false, a0.v, false, b0.v, (short)0, acc00, false, false);
        acc01 = __builtin_amdgcn_wmma_f32_16x16x32_bf16(false, a0.v, false, b1.v, (short)0, acc01, false, false);
        acc10 = __builtin_amdgcn_wmma_f32_16x16x32_bf16(false, a1.v, false, b0.v, (short)0, acc10, false, false);
        acc11 = __builtin_amdgcn_wmma_f32_16x16x32_bf16(false, a1.v, false, b1.v, (short)0, acc11, false, false);

        wait_stage();
        __syncthreads();
    }

    const int lcol = lane & 15;
    #pragma unroll
    for (int v = 0; v < 8; ++v) {
        const size_t row0 = blockRow + rowBase + v + 8 * half;       // C layout: VGPR v -> M=v / v+8
        const size_t row1 = row0 + 16;
        float* c0 = C + row0 * N + blockCol + colBase + lcol;
        float* c1 = C + row1 * N + blockCol + colBase + lcol;
        c0[0]  = acc00[v];
        c0[16] = acc01[v];
        c1[0]  = acc10[v];
        c1[16] = acc11[v];
    }
}

// ---------- conversion / setup kernels ----------
__global__ void cvt_f32_bf16(const float* __restrict__ s, unsigned short* __restrict__ d, int n) {
    int i = blockIdx.x * 256 + threadIdx.x;
    if (i < n) d[i] = f2bf(s[i]);
}

// w[K][N] (row-major) -> wT[N][K] bf16
__global__ void transpose_w_bf16(const float* __restrict__ w, unsigned short* __restrict__ wT,
                                 int K, int N) {
    int i = blockIdx.x * 256 + threadIdx.x;
    if (i >= K * N) return;
    int k = i / N, n = i - k * N;
    wT[(size_t)n * K + k] = f2bf(w[i]);
}

__global__ __launch_bounds__(256) void cb_norms(const float* __restrict__ cb, float* __restrict__ c2) {
    __shared__ float red[256];
    int j = blockIdx.x;
    float s = 0.f;
    for (int d = threadIdx.x; d < H2; d += 256) { float v = cb[(size_t)j * H2 + d]; s += v * v; }
    red[threadIdx.x] = s; __syncthreads();
    for (int o = 128; o > 0; o >>= 1) { if (threadIdx.x < o) red[threadIdx.x] += red[threadIdx.x + o]; __syncthreads(); }
    if (threadIdx.x == 0) c2[j] = red[0];
}

// ---------- fused bias + LayerNorm + ReLU ----------
__global__ __launch_bounds__(256) void ln_relu_kernel(
    const float* __restrict__ Y, const float* __restrict__ bias,
    const float* __restrict__ g, const float* __restrict__ b,
    unsigned short* __restrict__ obf, float* __restrict__ of32, int W)
{
    __shared__ float red[256];
    const int row = blockIdx.x;
    const float* y = Y + (size_t)row * W;
    float s = 0.f;
    for (int i = threadIdx.x; i < W; i += 256) s += y[i] + bias[i];
    red[threadIdx.x] = s; __syncthreads();
    for (int o = 128; o > 0; o >>= 1) { if (threadIdx.x < o) red[threadIdx.x] += red[threadIdx.x + o]; __syncthreads(); }
    float mu = red[0] / W; __syncthreads();
    float vs = 0.f;
    for (int i = threadIdx.x; i < W; i += 256) { float t = y[i] + bias[i] - mu; vs += t * t; }
    red[threadIdx.x] = vs; __syncthreads();
    for (int o = 128; o > 0; o >>= 1) { if (threadIdx.x < o) red[threadIdx.x] += red[threadIdx.x + o]; __syncthreads(); }
    float inv = rsqrtf(red[0] / W + 1e-5f);
    for (int i = threadIdx.x; i < W; i += 256) {
        float t = fmaxf((y[i] + bias[i] - mu) * inv * g[i] + b[i], 0.f);
        obf[(size_t)row * W + i] = f2bf(t);
        if (of32) of32[(size_t)row * W + i] = t;
    }
}

// ---------- VQ argmin over d2 = c2[j] - 2*G[i][j] ----------
__global__ __launch_bounds__(256) void vq_argmin_kernel(
    const float* __restrict__ G, const float* __restrict__ c2, int* __restrict__ idx)
{
    __shared__ float bv[256]; __shared__ int bi[256];
    const int row = blockIdx.x;
    float best = 3.4e38f; int bj = 0x7FFFFFFF;
    for (int j = threadIdx.x; j < KC; j += 256) {
        float d = c2[j] - 2.f * G[(size_t)row * KC + j];
        if (d < best || (d == best && j < bj)) { best = d; bj = j; }
    }
    bv[threadIdx.x] = best; bi[threadIdx.x] = bj; __syncthreads();
    for (int o = 128; o > 0; o >>= 1) {
        if (threadIdx.x < o) {
            float a = bv[threadIdx.x], c = bv[threadIdx.x + o];
            if (c < a || (c == a && bi[threadIdx.x + o] < bi[threadIdx.x])) {
                bv[threadIdx.x] = c; bi[threadIdx.x] = bi[threadIdx.x + o];
            }
        }
        __syncthreads();
    }
    if (threadIdx.x == 0) idx[row] = bi[0];
}

// ---------- gather quantized rows to bf16 ----------
__global__ __launch_bounds__(256) void gather_q_kernel(
    const int* __restrict__ idx, const float* __restrict__ cb, unsigned short* __restrict__ qbf)
{
    const int row = blockIdx.x;
    const float* c = cb + (size_t)idx[row] * H2;
    for (int d = threadIdx.x; d < H2; d += 256)
        qbf[(size_t)row * H2 + d] = f2bf(c[d]);
}

// ---------- pooled = mean over T of quantized ----------
__global__ __launch_bounds__(256) void pooled_kernel(
    const int* __restrict__ idx, const float* __restrict__ cb,
    unsigned short* __restrict__ pooledbf)
{
    __shared__ int sidx[TT];
    const int b = blockIdx.x;
    for (int t = threadIdx.x; t < TT; t += 256) sidx[t] = idx[b * TT + t];
    __syncthreads();
    for (int d = threadIdx.x; d < H2; d += 256) {
        float s = 0.f;
        for (int t = 0; t < TT; ++t) s += cb[(size_t)sidx[t] * H2 + d];
        pooledbf[(size_t)b * H2 + d] = f2bf(s * (1.f / TT));
    }
}

// ---------- per-row bias + ReLU -> bf16 (any width) ----------
__global__ __launch_bounds__(256) void bias_relu_rows(
    const float* __restrict__ Y, const float* __restrict__ bias,
    unsigned short* __restrict__ out, int W)
{
    const size_t base = (size_t)blockIdx.x * W;
    for (int i = threadIdx.x; i < W; i += 256)
        out[base + i] = f2bf(fmaxf(Y[base + i] + bias[i], 0.f));
}

// ---------- loss reductions ----------
__global__ void init_acc_kernel(float* acc) {
    if (threadIdx.x < 8) acc[threadIdx.x] = 0.f;
}

__global__ __launch_bounds__(256) void recon_loss_kernel(
    const float* __restrict__ Y, const float* __restrict__ bias,
    const float* __restrict__ x, float* __restrict__ acc)
{
    __shared__ float red[256];
    float s = 0.f;
    const int total = MM * DD;
    for (int i = blockIdx.x * 256 + threadIdx.x; i < total; i += gridDim.x * 256) {
        float d = Y[i] + bias[i & (DD - 1)] - x[i];
        s += d * d;
    }
    red[threadIdx.x] = s; __syncthreads();
    for (int o = 128; o > 0; o >>= 1) { if (threadIdx.x < o) red[threadIdx.x] += red[threadIdx.x + o]; __syncthreads(); }
    if (threadIdx.x == 0) atomicAdd(&acc[0], red[0]);
}

__global__ __launch_bounds__(256) void commit_loss_kernel(
    const float* __restrict__ enc, const int* __restrict__ idx,
    const float* __restrict__ cb, float* __restrict__ acc)
{
    __shared__ float red[256];
    float s = 0.f;
    const int total = MM * H2;
    for (int i = blockIdx.x * 256 + threadIdx.x; i < total; i += gridDim.x * 256) {
        int row = i >> 9;            // H2 = 512
        int d   = i & (H2 - 1);
        float t = enc[i] - cb[(size_t)idx[row] * H2 + d];
        s += t * t;
    }
    red[threadIdx.x] = s; __syncthreads();
    for (int o = 128; o > 0; o >>= 1) { if (threadIdx.x < o) red[threadIdx.x] += red[threadIdx.x + o]; __syncthreads(); }
    if (threadIdx.x == 0) atomicAdd(&acc[1], red[0]);
}

// ---------- projection tail: bias + L2 normalize ----------
__global__ __launch_bounds__(128) void bias_norm_kernel(
    const float* __restrict__ Y, const float* __restrict__ bias, float* __restrict__ np)
{
    __shared__ float red[128];
    const int row = blockIdx.x;
    const int t = threadIdx.x;
    float v = Y[(size_t)row * P2 + t] + bias[t];
    red[t] = v * v; __syncthreads();
    for (int o = 64; o > 0; o >>= 1) { if (t < o) red[t] += red[t + o]; __syncthreads(); }
    float inv = 1.f / fmaxf(sqrtf(red[0]), 1e-12f);
    np[(size_t)row * P2 + t] = v * inv;
}

// ---------- contrastive loss (single block, streaming LSE) ----------
__global__ void contrastive_kernel(const float* __restrict__ np, float* __restrict__ acc)
{
    __shared__ float sp[MB * P2];
    __shared__ float red[MB];
    for (int i = threadIdx.x; i < MB * P2; i += blockDim.x) sp[i] = np[i];
    __syncthreads();
    const int i = threadIdx.x;
    if (i < MB) {
        float m = -3.4e38f, s = 0.f, diag = 0.f;
        for (int j = 0; j < MB; ++j) {
            float d = 0.f;
            for (int k = 0; k < P2; ++k) d += sp[i * P2 + k] * sp[j * P2 + k];
            d *= 10.f;                       // 1 / TEMP
            if (j == i) diag = d;
            if (d > m) { s = s * __expf(m - d) + 1.f; m = d; }
            else       { s += __expf(d - m); }
        }
        red[i] = (m + __logf(s)) - diag;
    }
    __syncthreads();
    if (threadIdx.x == 0) {
        float t = 0.f;
        for (int j = 0; j < MB; ++j) t += red[j];
        acc[2] = t / MB;
    }
}

// ---------- finalize ----------
__global__ void finalize_kernel(const int* __restrict__ idx, const float* __restrict__ acc,
                                float* __restrict__ out)
{
    int i = blockIdx.x * 256 + threadIdx.x;
    if (i < MM) out[i] = (float)idx[i];
    if (blockIdx.x == 0 && threadIdx.x == 0) {
        float recon  = acc[0] / (float)(MM * DD);
        float commit = acc[1] / (float)(MM * H2);
        float codebk = commit;
        float contra = acc[2];
        float total  = recon + commit + 0.25f * codebk + 0.5f * contra;
        out[MM + 0] = total;
        out[MM + 1] = recon;
        out[MM + 2] = commit;
        out[MM + 3] = codebk;
        out[MM + 4] = contra;
    }
}

// ---------- launch ----------
extern "C" void kernel_launch(void* const* d_in, const int* in_sizes, int n_in,
                              void* d_out, int out_size, void* d_ws, size_t ws_size,
                              hipStream_t stream)
{
    const float* x       = (const float*)d_in[0];
    const float* enc_w1  = (const float*)d_in[1];
    const float* enc_b1  = (const float*)d_in[2];
    const float* ln1_g   = (const float*)d_in[3];
    const float* ln1_b   = (const float*)d_in[4];
    const float* enc_w2  = (const float*)d_in[5];
    const float* enc_b2  = (const float*)d_in[6];
    const float* ln2_g   = (const float*)d_in[7];
    const float* ln2_b   = (const float*)d_in[8];
    const float* cb      = (const float*)d_in[9];
    const float* dec_w1  = (const float*)d_in[10];
    const float* dec_b1  = (const float*)d_in[11];
    const float* dec_w2  = (const float*)d_in[12];
    const float* dec_b2  = (const float*)d_in[13];
    const float* proj_w1 = (const float*)d_in[14];
    const float* proj_b1 = (const float*)d_in[15];
    const float* proj_w2 = (const float*)d_in[16];
    const float* proj_b2 = (const float*)d_in[17];
    float* out = (float*)d_out;

    // workspace layout
    char* ws = (char*)d_ws;
    size_t o = 0;
    auto alloc = [&](size_t bytes) { size_t r = o; o = (o + bytes + 255) & ~(size_t)255; return r; };
    unsigned short* xbf   = (unsigned short*)(ws + alloc((size_t)MM * DD * 2));
    unsigned short* w1T   = (unsigned short*)(ws + alloc((size_t)DD * H1 * 2));   // [H1][DD]
    unsigned short* w2T   = (unsigned short*)(ws + alloc((size_t)H1 * H2 * 2));   // [H2][H1]
    unsigned short* cbbf  = (unsigned short*)(ws + alloc((size_t)KC * H2 * 2));   // [KC][H2] = BT for VQ
    unsigned short* dw1T  = (unsigned short*)(ws + alloc((size_t)H2 * H1 * 2));   // [H1][H2]
    unsigned short* dw2T  = (unsigned short*)(ws + alloc((size_t)H1 * DD * 2));   // [DD][H1]
    unsigned short* pw1T  = (unsigned short*)(ws + alloc((size_t)H2 * P1 * 2));   // [P1][H2]
    unsigned short* pw2T  = (unsigned short*)(ws + alloc((size_t)P1 * P2 * 2));   // [P2][P1]
    unsigned short* hbf   = (unsigned short*)(ws + alloc((size_t)MM * H1 * 2));   // reused as dh
    unsigned short* encbf = (unsigned short*)(ws + alloc((size_t)MM * H2 * 2));   // reused as q
    float* Ybuf   = (float*)(ws + alloc((size_t)MM * DD * 4));                    // big GEMM scratch
    float* encf   = (float*)(ws + alloc((size_t)MM * H2 * 4));
    int*   idx    = (int*)  (ws + alloc((size_t)MM * 4));
    float* c2     = (float*)(ws + alloc((size_t)KC * 4));
    unsigned short* pooledbf = (unsigned short*)(ws + alloc((size_t)128 * H2 * 2)); // padded to 128 rows
    unsigned short* p1bf     = (unsigned short*)(ws + alloc((size_t)128 * P1 * 2)); // padded to 128 rows
    float* nproj  = (float*)(ws + alloc((size_t)MB * P2 * 4));
    float* acc    = (float*)(ws + alloc(8 * 4));

    // --- conversions / transposes (weights are B matrices; store N-major) ---
    cvt_f32_bf16<<<(MM * DD) / 256, 256, 0, stream>>>(x, xbf, MM * DD);
    transpose_w_bf16<<<(DD * H1) / 256, 256, 0, stream>>>(enc_w1, w1T, DD, H1);
    transpose_w_bf16<<<(H1 * H2) / 256, 256, 0, stream>>>(enc_w2, w2T, H1, H2);
    cvt_f32_bf16<<<(KC * H2) / 256, 256, 0, stream>>>(cb, cbbf, KC * H2);
    transpose_w_bf16<<<(H2 * H1) / 256, 256, 0, stream>>>(dec_w1, dw1T, H2, H1);
    transpose_w_bf16<<<(H1 * DD) / 256, 256, 0, stream>>>(dec_w2, dw2T, H1, DD);
    transpose_w_bf16<<<(H2 * P1) / 256, 256, 0, stream>>>(proj_w1, pw1T, H2, P1);
    transpose_w_bf16<<<(P1 * P2) / 256, 256, 0, stream>>>(proj_w2, pw2T, P1, P2);
    cb_norms<<<KC, 256, 0, stream>>>(cb, c2);
    init_acc_kernel<<<1, 32, 0, stream>>>(acc);

    // --- encoder ---
    gemm_bf16_wmma<<<dim3(H1 / BN, MM / BM), 256, 0, stream>>>(xbf, w1T, Ybuf, MM, H1, DD);
    ln_relu_kernel<<<MM, 256, 0, stream>>>(Ybuf, enc_b1, ln1_g, ln1_b, hbf, nullptr, H1);
    gemm_bf16_wmma<<<dim3(H2 / BN, MM / BM), 256, 0, stream>>>(hbf, w2T, Ybuf, MM, H2, H1);
    ln_relu_kernel<<<MM, 256, 0, stream>>>(Ybuf, enc_b2, ln2_g, ln2_b, encbf, encf, H2);

    // --- VQ ---
    gemm_bf16_wmma<<<dim3(KC / BN, MM / BM), 256, 0, stream>>>(encbf, cbbf, Ybuf, MM, KC, H2);
    vq_argmin_kernel<<<MM, 256, 0, stream>>>(Ybuf, c2, idx);
    gather_q_kernel<<<MM, 256, 0, stream>>>(idx, cb, encbf);      // encbf reused as q_bf
    pooled_kernel<<<MB, 256, 0, stream>>>(idx, cb, pooledbf);

    // --- decoder ---
    gemm_bf16_wmma<<<dim3(H1 / BN, MM / BM), 256, 0, stream>>>(encbf, dw1T, Ybuf, MM, H1, H2);
    bias_relu_rows<<<MM, 256, 0, stream>>>(Ybuf, dec_b1, hbf, H1);  // hbf reused as dh_bf
    gemm_bf16_wmma<<<dim3(DD / BN, MM / BM), 256, 0, stream>>>(hbf, dw2T, Ybuf, MM, DD, H1);

    // --- losses on big tensors ---
    recon_loss_kernel<<<4096, 256, 0, stream>>>(Ybuf, dec_b2, x, acc);
    commit_loss_kernel<<<4096, 256, 0, stream>>>(encf, idx, cb, acc);

    // --- projection head (M padded 64 -> 128; pad rows never read) ---
    gemm_bf16_wmma<<<dim3(P1 / BN, 1), 256, 0, stream>>>(pooledbf, pw1T, Ybuf, 128, P1, H2);
    bias_relu_rows<<<MB, 256, 0, stream>>>(Ybuf, proj_b1, p1bf, P1);
    gemm_bf16_wmma<<<dim3(P2 / BN, 1), 256, 0, stream>>>(p1bf, pw2T, Ybuf, 128, P2, P1);
    bias_norm_kernel<<<MB, 128, 0, stream>>>(Ybuf, proj_b2, nproj);
    contrastive_kernel<<<1, 128, 0, stream>>>(nproj, acc);

    // --- output: indices then 5 scalars ---
    finalize_kernel<<<MM / 256, 256, 0, stream>>>(idx, acc, out);
    (void)in_sizes; (void)n_in; (void)out_size; (void)ws_size;
}
